// StereoContrastiveLoss_52716428591128
// MI455X (gfx1250) — matrix operand order repaired and checked by reading, not compiled, not run on hardware
//
#include <hip/hip_runtime.h>
#include <hip/hip_fp16.h>

typedef __attribute__((ext_vector_type(16))) _Float16 v16h;
typedef __attribute__((ext_vector_type(8)))  _Float16 v8h;
typedef __attribute__((ext_vector_type(8)))  float    v8f;

#define B_    2
#define C_    128
#define H_    64
#define W_    128
#define HW    (H_*W_)        // 8192
#define NPIX  (B_*H_*W_)     // 16384
#define HI_   256
#define WI_   512
#define NQ    6000
#define NNEG  60
#define INV_TEMP (1.0f/0.07f)

// ---------------- helpers ----------------

__device__ __forceinline__ float wred32(float v) {
  // full wave32 sum, result broadcast to all lanes
  v += __shfl_xor(v, 16, 32);
  v += __shfl_xor(v,  8, 32);
  v += __shfl_xor(v,  4, 32);
  v += __shfl_xor(v,  2, 32);
  v += __shfl_xor(v,  1, 32);
  return v;
}

__device__ __forceinline__ unsigned hashu(unsigned x) {
  x ^= x >> 16; x *= 0x7feb352du;
  x ^= x >> 15; x *= 0x846ca68bu;
  x ^= x >> 16;
  return x;
}

struct Bil { int x0, x1, y0, y1; float wx, wy; };

__device__ __forceinline__ Bil mkbil(float ix, float iy, int Wd, int Hd) {
  ix = fminf(fmaxf(ix, 0.0f), (float)(Wd - 1));
  iy = fminf(fmaxf(iy, 0.0f), (float)(Hd - 1));
  Bil b;
  float x0f = floorf(ix), y0f = floorf(iy);
  b.wx = ix - x0f; b.wy = iy - y0f;
  b.x0 = (int)x0f; b.y0 = (int)y0f;
  b.x1 = b.x0 + 1; if (b.x1 > Wd - 1) b.x1 = Wd - 1;
  b.y1 = b.y0 + 1; if (b.y1 > Hd - 1) b.y1 = Hd - 1;
  return b;
}

__device__ __forceinline__ float samp(const float* __restrict__ p, const Bil& b) {
  float v00 = p[b.y0 * W_ + b.x0], v01 = p[b.y0 * W_ + b.x1];
  float v10 = p[b.y1 * W_ + b.x0], v11 = p[b.y1 * W_ + b.x1];
  return (v00 * (1.0f - b.wx) + v01 * b.wx) * (1.0f - b.wy)
       + (v10 * (1.0f - b.wx) + v11 * b.wx) * b.wy;
}

// ---------------- kernel 1: normalize queue columns, transpose to f16 [NQ][C] ----------------

__global__ __launch_bounds__(256) void k_queue(const float* __restrict__ queue,
                                               _Float16* __restrict__ qnT) {
  int k = blockIdx.x * 256 + threadIdx.x;
  if (k >= NQ) return;
  float n2 = 0.0f;
  for (int ci = 0; ci < C_; ++ci) {
    float v = queue[ci * NQ + k];
    n2 += v * v;
  }
  float inv = 1.0f / fmaxf(sqrtf(n2), 1e-12f);
  for (int ci = 0; ci < C_; ++ci)
    qnT[(size_t)k * C_ + ci] = (_Float16)(queue[ci * NQ + k] * inv);
}

// ---------------- kernel 2: tmp1 = warp(index, disps_r) ----------------

__global__ __launch_bounds__(256) void k_warp_index(const float* __restrict__ tr,
                                                    float* __restrict__ tmp1) {
  int pix = blockIdx.x * 256 + threadIdx.x;   // exact: 64*256 = 16384
  int bi = pix / HW;
  int rem = pix - bi * HW;
  int y = rem >> 7, x = rem & 127;
  float dr = tr[bi * HI_ * WI_ + (y * 4) * WI_ + x * 4] * 0.25f;
  float xb = (float)x * (1.0f / 127.0f);
  // ix = (xb + dr/w)*w - 0.5, clipped (y-interp is a no-op for the index image)
  float ix = fminf(fmaxf((xb + dr * (1.0f / 128.0f)) * 128.0f - 0.5f, 0.0f), 127.0f);
  float x0f = floorf(ix);
  float wx = ix - x0f;
  float x1f = fminf(x0f + 1.0f, 127.0f);
  tmp1[pix] = x0f * (1.0f - wx) + x1f * wx;
}

// ---------------- kernel 3: per-pixel query/positive/negatives (one wave32 per pixel) ----------------

__global__ __launch_bounds__(256) void k_pixel(const float* __restrict__ ref,
                                               const float* __restrict__ tgt,
                                               const float* __restrict__ tl,
                                               const float* __restrict__ tmp1,
                                               _Float16* __restrict__ wsQ,
                                               float* __restrict__ maskArr,
                                               float* __restrict__ rowpart,
                                               float* __restrict__ lposArr) {
  const int wave = threadIdx.x >> 5;
  const int lane = threadIdx.x & 31;
  const int pix  = blockIdx.x * 8 + wave;     // exact: 2048*8 = 16384
  const int bi = pix / HW;
  const int rem = pix - bi * HW;
  const int y = rem >> 7, x = rem & 127;

  const float dl = tl[bi * HI_ * WI_ + (y * 4) * WI_ + x * 4] * 0.25f;
  const float xb = (float)x * (1.0f / 127.0f);
  const float yb = (float)y * (1.0f / 63.0f);

  // -- query: channels lane + 32*i --
  const float* refp = ref + (size_t)bi * C_ * HW + y * W_ + x;
  float q[4], n2 = 0.0f;
#pragma unroll
  for (int i = 0; i < 4; ++i) {
    q[i] = refp[(size_t)(lane + 32 * i) * HW];
    n2 += q[i] * q[i];
  }
  n2 = wred32(n2);
  float inv = 1.0f / fmaxf(sqrtf(n2), 1e-12f);
#pragma unroll
  for (int i = 0; i < 4; ++i) {
    q[i] *= inv;
    wsQ[(size_t)pix * C_ + lane + 32 * i] = (_Float16)q[i];
  }

  // -- positive sample at (xb - dl/w, yb) --
  const float pxn = xb - dl * (1.0f / 128.0f);
  Bil bp = mkbil(pxn * 128.0f - 0.5f, yb * 64.0f - 0.5f, W_, H_);
  const float* tgtb = tgt + (size_t)bi * C_ * HW;
  float p4[4], pn2 = 0.0f, dotp = 0.0f;
#pragma unroll
  for (int i = 0; i < 4; ++i) {
    p4[i] = samp(tgtb + (size_t)(lane + 32 * i) * HW, bp);
    pn2 += p4[i] * p4[i];
    dotp += q[i] * p4[i];
  }
  pn2  = wred32(pn2);
  dotp = wred32(dotp) * (1.0f / fmaxf(sqrtf(pn2), 1e-12f));

  // -- occlusion mask (sample tmp1 at the same warped coordinate) --
  float l2r2l = samp(tmp1 + bi * HW, bp);
  float mv = ((fabsf((float)x - l2r2l) < 3.0f) && (dl > 0.0f) && (pxn >= 0.0f)) ? 1.0f : 0.0f;
  const float scale = mv * INV_TEMP;
  const float lp = dotp * scale;
  float esum = __expf(lp);

  // -- 60 negatives --
  const float xr = pxn * 128.0f;
#pragma unroll 4
  for (int j = 0; j < NNEG; ++j) {
    unsigned id = (unsigned)(pix * NNEG + j);
    unsigned h1 = hashu(id ^ 0x9E3779B9u);
    unsigned h2 = hashu(id + 0x85EBCA6Bu);
    float mag = 1.0f + (float)(h1 % 24u);          // randint [1,25)
    float sg  = (h2 & 1u) ? 1.0f : -1.0f;          // sign(uniform-0.5)
    float fx = fmodf(xr + mag * sg, 128.0f);
    if (fx < 0.0f) fx += 128.0f;
    fx *= (1.0f / 128.0f);
    // ix = fx*w - 0.5 ; fy = fx/h  ->  iy = fy*h - 0.5 = fx - 0.5
    Bil bn = mkbil(fx * 128.0f - 0.5f, fx - 0.5f, W_, H_);
    float nn2 = 0.0f, dt = 0.0f;
#pragma unroll
    for (int i = 0; i < 4; ++i) {
      float nv = samp(tgtb + (size_t)(lane + 32 * i) * HW, bn);
      nn2 += nv * nv;
      dt  += q[i] * nv;
    }
    nn2 = wred32(nn2);
    dt  = wred32(dt);
    esum += __expf(dt * (1.0f / fmaxf(sqrtf(nn2), 1e-12f)) * scale);
  }

  if (lane == 0) {
    rowpart[pix] = esum;   // exp(l_pos) + sum_j exp(l_neg_j)  (logits already masked+scaled)
    lposArr[pix] = lp;
    maskArr[pix] = mv;
  }
}

// ---------------- kernel 4: WMMA GEMM (query f16 @ qnT) fused with exp-sum ----------------
// One wave per 16-row strip; loops over all 375 column tiles of 16; K=128 -> 4 WMMAs/tile.

__global__ __launch_bounds__(256) void k_gemm_softmax(const _Float16* __restrict__ A,
                                                      const _Float16* __restrict__ Bq,
                                                      const float* __restrict__ maskArr,
                                                      float* __restrict__ gemmsum) {
  const int lane   = threadIdx.x & 31;
  const int wave   = (blockIdx.x * 256 + threadIdx.x) >> 5;  // 1024 waves exactly
  const int rstart = wave * 16;
  const int half   = lane >> 4;
  const int m      = lane & 15;

  // A-matrix 16x32 f16 lane layout (ISA 7.12.2): lane m holds row M=m;
  // elems 0..7 -> K = half*8 + 0..7 ; elems 8..15 -> K = 16 + half*8 + 0..7
  v16h at[4];
  const _Float16* arow = A + (size_t)(rstart + m) * C_;
#pragma unroll
  for (int t = 0; t < 4; ++t) {
    v8h lo = *reinterpret_cast<const v8h*>(arow + t * 32 + half * 8);
    v8h hi = *reinterpret_cast<const v8h*>(arow + t * 32 + 16 + half * 8);
#pragma unroll
    for (int j = 0; j < 8; ++j) { at[t][j] = lo[j]; at[t][8 + j] = hi[j]; }
  }

  // C/D layout: VGPR v -> row = v + 8*half, col = lane&15
  float srow[8], esum[8];
#pragma unroll
  for (int v = 0; v < 8; ++v) {
    srow[v] = maskArr[rstart + v + 8 * half] * INV_TEMP;
    esum[v] = 0.0f;
  }

  for (int nt = 0; nt < NQ / 16; ++nt) {
    const int gcol = nt * 16 + m;
    const _Float16* brow = Bq + (size_t)gcol * C_;
    if (nt + 1 < NQ / 16) __builtin_prefetch(brow + 16 * C_, 0, 1);
    v8f acc = {};
#pragma unroll
    for (int t = 0; t < 4; ++t) {
      // B-matrix 32x16 f16: lane n=lane&15 holds col n; elems 0..7 -> K = half*8 + 0..7 ; elems 8..15 -> K = 16 + half*8 + 0..7
      v8h blo = *reinterpret_cast<const v8h*>(brow + t * 32 + half * 8);
      v8h bhi = *reinterpret_cast<const v8h*>(brow + t * 32 + 16 + half * 8);
      v16h bt;
#pragma unroll
      for (int j = 0; j < 8; ++j) { bt[j] = blo[j]; bt[8 + j] = bhi[j]; }
      acc = __builtin_amdgcn_wmma_f32_16x16x32_f16(
          /*neg_a=*/false, at[t], /*neg_b=*/false, bt,
          /*c_mod=*/(short)0, acc, /*reuse_a=*/false, /*reuse_b=*/false);
    }
#pragma unroll
    for (int v = 0; v < 8; ++v)
      esum[v] += __expf(acc[v] * srow[v]);   // |logit| <= 1/0.07: no overflow, no max pass
  }

  // reduce over the 16 columns (lanes within each half)
#pragma unroll
  for (int v = 0; v < 8; ++v) {
    float e = esum[v];
    e += __shfl_xor(e, 1, 32);
    e += __shfl_xor(e, 2, 32);
    e += __shfl_xor(e, 4, 32);
    e += __shfl_xor(e, 8, 32);
    if (m == 0) gemmsum[rstart + v + 8 * half] = e;
  }
}

// ---------------- kernel 5: final loss reduction ----------------

__global__ __launch_bounds__(256) void k_loss(const float* __restrict__ rowpart,
                                              const float* __restrict__ gemmsum,
                                              const float* __restrict__ lposArr,
                                              float* __restrict__ out) {
  __shared__ float red[256];
  float acc = 0.0f;
  for (int r = threadIdx.x; r < NPIX; r += 256)
    acc += logf(rowpart[r] + gemmsum[r]) - lposArr[r];
  red[threadIdx.x] = acc;
  __syncthreads();
  for (int s = 128; s > 0; s >>= 1) {
    if (threadIdx.x < (unsigned)s) red[threadIdx.x] += red[threadIdx.x + s];
    __syncthreads();
  }
  if (threadIdx.x == 0) out[0] = red[0] / (float)NPIX;
}

// ---------------- launcher ----------------

extern "C" void kernel_launch(void* const* d_in, const int* in_sizes, int n_in,
                              void* d_out, int out_size, void* d_ws, size_t ws_size,
                              hipStream_t stream) {
  (void)in_sizes; (void)n_in; (void)out_size; (void)ws_size;
  const float* ref = (const float*)d_in[0];
  const float* tgt = (const float*)d_in[1];
  const float* tl  = (const float*)d_in[2];
  const float* tr  = (const float*)d_in[3];
  const float* qu  = (const float*)d_in[4];

  char* ws = (char*)d_ws;
  const size_t QNT_BYTES = (size_t)NQ * C_ * sizeof(_Float16);     // 1,536,000 (16B-aligned)
  const size_t WSQ_BYTES = (size_t)NPIX * C_ * sizeof(_Float16);   // 4,194,304
  _Float16* qnT     = (_Float16*)(ws);
  _Float16* wsQ     = (_Float16*)(ws + QNT_BYTES);
  float*    tmp1    = (float*)(ws + QNT_BYTES + WSQ_BYTES);
  float*    maskArr = tmp1 + NPIX;
  float*    rowpart = maskArr + NPIX;
  float*    lposArr = rowpart + NPIX;
  float*    gemmsum = lposArr + NPIX;

  k_queue<<<(NQ + 255) / 256, 256, 0, stream>>>(qu, qnT);
  k_warp_index<<<NPIX / 256, 256, 0, stream>>>(tr, tmp1);
  k_pixel<<<NPIX / 8, 256, 0, stream>>>(ref, tgt, tl, tmp1, wsQ, maskArr, rowpart, lposArr);
  k_gemm_softmax<<<(NPIX / 16) / 8, 256, 0, stream>>>(wsQ, qnT, maskArr, gemmsum);
  k_loss<<<1, 256, 0, stream>>>(rowpart, gemmsum, lposArr, (float*)d_out);
}
